// MultiheadedAttentionMechanism_70583492542683
// MI455X (gfx1250) — matrix operand order repaired and checked by reading, not compiled, not run on hardware
//
#include <hip/hip_runtime.h>
#include <hip/hip_bf16.h>

typedef __bf16 bf16_t;
typedef __attribute__((ext_vector_type(16))) __bf16 v16bf;
typedef __attribute__((ext_vector_type(8)))  __bf16 v8bf;
typedef __attribute__((ext_vector_type(8)))  float  v8f;

#define BATCH 8
#define HEADS 16
#define SEQ   1024
#define DIM   1024
#define DK    64

// GEMM block tile
#define BM 128
#define BN 128
#define BK 32

// ---------- helpers ----------

__device__ inline bf16_t f2bf(float x) {
    unsigned u = __float_as_uint(x);
    unsigned r = (u + 0x7FFFu + ((u >> 16) & 1u)) >> 16;
    unsigned short s = (unsigned short)r;
    return __builtin_bit_cast(bf16_t, s);
}

// A-fragment (16x32 bf16, row-major source, row stride ld elements).
// Layout: lanes 0-15 hold M=0-15 / K 0..15;
// lanes 16-31 hold M=0-15 / K 16..31.
__device__ inline v16bf load_a_frag(const bf16_t* base, int ld) {
    int lane = threadIdx.x & 31;
    int m = lane & 15, half = lane >> 4;
    return *(const v16bf*)(base + (size_t)m * ld + half * 16);
}

// B-fragment (32x16 bf16) read from a transposed (N-major) buffer:
// Bt[n][k], row stride ld. Lane n = lane&15, K-half = lane>>4 -> 16
// contiguous K values per lane = one 32-byte load.
__device__ inline v16bf load_b_frag(const bf16_t* base, int ld) {
    int lane = threadIdx.x & 31;
    int n = lane & 15, half = lane >> 4;
    return *(const v16bf*)(base + (size_t)n * ld + half * 16);
}

__device__ inline v8f wmma_bf16(v16bf a, v16bf b, v8f c) {
    return __builtin_amdgcn_wmma_f32_16x16x32_bf16(
        /*neg_a=*/false, a, /*neg_b=*/false, b,
        /*c_mod=*/(short)0, c, /*reuse_a=*/false, /*reuse_b=*/false);
}

// Async global->LDS copy of 16 bytes per lane (CDNA5, ASYNCcnt-tracked).
// lds_off = wave-relative LDS byte offset (low 32 bits of flat shared ptr).
__device__ inline void async_copy16(unsigned lds_off, const void* gaddr) {
    asm volatile("global_load_async_to_lds_b128 %0, %1, off"
                 :: "v"(lds_off), "v"(gaddr) : "memory");
}

__device__ inline void wait_async0() {
    asm volatile("s_wait_asynccnt 0x0" ::: "memory");
}

// ---------- conversion kernels ----------

__global__ void cvt_f32_bf16_kernel(const float* __restrict__ src,
                                    bf16_t* __restrict__ dst, int n) {
    int i = blockIdx.x * blockDim.x + threadIdx.x;
    if (i < n) dst[i] = f2bf(src[i]);
}

// dst[n*K + k] = cvt(src[k*N + n])   (K = N = 1024 here)
__global__ void transpose_cvt_kernel(const float* __restrict__ src,
                                     bf16_t* __restrict__ dst) {
    int i = blockIdx.x * blockDim.x + threadIdx.x;
    int k = i >> 10, n = i & 1023;
    dst[n * DIM + k] = f2bf(src[i]);
}

// ---------- LDS-tiled bf16 GEMM: C = A[MxK] * Bt[NxK]^T + bias ----------
// 256 threads = 8 waves (4x2). Block tile 128x128, K-panel 32.
// Each wave: 32x64 register tile = 2x4 WMMA accumulators.
// Panels staged with global_load_async_to_lds_b128.
// mode 0: bf16 row-major out          (Q, K projections)
// mode 1: bf16 head-transposed out    (V projection -> [B,H,64,SEQ])
// mode 2: fp32 row-major out          (final out projection)

__global__ __launch_bounds__(256) void gemm_bf16_tiled_kernel(
    const bf16_t* __restrict__ A, const bf16_t* __restrict__ Bt,
    const float* __restrict__ bias, float* __restrict__ outf,
    bf16_t* __restrict__ outb, int M, int N, int K, int mode) {
    __shared__ __align__(64) bf16_t lA[BM * BK];   // [m][k], 8 KB
    __shared__ __align__(64) bf16_t lB[BN * BK];   // [n][k], 8 KB

    int tid  = threadIdx.x;
    int wave = tid >> 5, lane = tid & 31;
    int tiles_n = N / BN;
    int tm0 = (blockIdx.x / tiles_n) * BM;
    int tn0 = (blockIdx.x % tiles_n) * BN;
    int wm = wave >> 1, wn = wave & 1;             // 4x2 wave grid

    unsigned ldsA = (unsigned)(size_t)(void*)lA;
    unsigned ldsB = (unsigned)(size_t)(void*)lB;

    v8f zero = {};
    v8f acc[2][4];
#pragma unroll
    for (int i = 0; i < 2; ++i)
#pragma unroll
        for (int j = 0; j < 4; ++j) acc[i][j] = zero;

    const bf16_t* Ab = A  + (size_t)tm0 * K;
    const bf16_t* Bb = Bt + (size_t)tn0 * K;

    for (int k0 = 0; k0 < K; k0 += BK) {
        // stage panels: 512 16-byte chunks each; thread does chunks tid, tid+256
#pragma unroll
        for (int s = 0; s < 2; ++s) {
            int c = tid + s * 256;
            int mrow = c >> 2;
            int koff = (c & 3) * 8;
            async_copy16(ldsA + (unsigned)c * 16u,
                         (const void*)(Ab + (size_t)mrow * K + k0 + koff));
            async_copy16(ldsB + (unsigned)c * 16u,
                         (const void*)(Bb + (size_t)mrow * K + k0 + koff));
        }
        wait_async0();
        __syncthreads();

        v16bf af[2], bfr[4];
#pragma unroll
        for (int i = 0; i < 2; ++i)
            af[i] = load_a_frag(lA + (wm * 32 + i * 16) * BK, BK);
#pragma unroll
        for (int j = 0; j < 4; ++j)
            bfr[j] = load_b_frag(lB + (wn * 64 + j * 16) * BK, BK);
#pragma unroll
        for (int i = 0; i < 2; ++i)
#pragma unroll
            for (int j = 0; j < 4; ++j)
                acc[i][j] = wmma_bf16(af[i], bfr[j], acc[i][j]);
        __syncthreads();
    }

    // epilogue
    int half = lane >> 4;
#pragma unroll
    for (int i = 0; i < 2; ++i) {
#pragma unroll
        for (int j = 0; j < 4; ++j) {
            int n = tn0 + wn * 64 + j * 16 + (lane & 15);
            float bv = bias ? bias[n] : 0.0f;
#pragma unroll
            for (int r = 0; r < 8; ++r) {
                int m = tm0 + wm * 32 + i * 16 + r + half * 8;
                float v = acc[i][j][r] + bv;
                if (mode == 2) {
                    outf[(size_t)m * N + n] = v;
                } else if (mode == 0) {
                    outb[(size_t)m * N + n] = f2bf(v);
                } else { // mode 1: [B][H][DK][SEQ]
                    int b_ = m >> 10, l = m & 1023;
                    int h_ = n >> 6,  cc = n & 63;
                    outb[((size_t)((b_ * HEADS + h_) * DK + cc) << 10) + l] = f2bf(v);
                }
            }
        }
    }
}

// ---------- fused attention: scores -> softmax -> P*V ----------
// One block per (b, h, 16-row q-tile). 256 threads = 8 waves.

__global__ __launch_bounds__(256) void attn_fused_kernel(
    const bf16_t* __restrict__ Q,    // [B*SEQ, DIM] bf16
    const bf16_t* __restrict__ Kb,   // [B*SEQ, DIM] bf16
    const bf16_t* __restrict__ Vt,   // [B, H, DK, SEQ] bf16
    float* __restrict__ attn,        // [B, H, SEQ, SEQ] fp32 (d_out region)
    bf16_t* __restrict__ ctxb) {     // [B*SEQ, DIM] bf16
    __shared__ __align__(64) bf16_t P[16][SEQ];   // 32 KB probs
    __shared__ float red[16][16];

    int blk = blockIdx.x;          // b*H*64 + h*64 + qt
    int qt = blk & 63;
    int h  = (blk >> 6) & 15;
    int b  = blk >> 10;
    int wave = threadIdx.x >> 5, lane = threadIdx.x & 31;
    int q0 = qt * 16;

    const bf16_t* Qbase = Q  + ((size_t)(b * SEQ + q0)) * DIM + h * DK;
    const bf16_t* Kbase = Kb + ((size_t)(b * SEQ)) * DIM + h * DK;
    float* Srow = attn + ((size_t)((b * HEADS + h) * SEQ + q0)) * SEQ;
    const float scale = 0.125f;    // 1/sqrt(64)

    // phase 1: S = (Q Kt) * scale  -> global (L2-hot)
    for (int nb = wave; nb < 64; nb += 8) {
        v8f c = {};
#pragma unroll
        for (int k0 = 0; k0 < DK; k0 += 32) {
            v16bf a  = load_a_frag(Qbase + k0, DIM);
            v16bf bb = load_b_frag(Kbase + (size_t)(nb * 16) * DIM + k0, DIM);
            c = wmma_bf16(a, bb, c);
        }
        int n = nb * 16 + (lane & 15);
        int half = lane >> 4;
#pragma unroll
        for (int r = 0; r < 8; ++r)
            Srow[(size_t)(r + half * 8) * SEQ + n] = c[r] * scale;
    }
    __threadfence_block();
    __syncthreads();

    // phase 2: row softmax over 1024 keys; 16 threads per row
    int row = threadIdx.x >> 4;
    int tp  = threadIdx.x & 15;
    float* S = Srow + (size_t)row * SEQ;
    float mx = -3.0e38f;
    for (int cidx = tp; cidx < SEQ; cidx += 16) mx = fmaxf(mx, S[cidx]);
    red[row][tp] = mx;
    __syncthreads();
    mx = red[row][0];
#pragma unroll
    for (int i = 1; i < 16; ++i) mx = fmaxf(mx, red[row][i]);
    float sum = 0.0f;
    for (int cidx = tp; cidx < SEQ; cidx += 16) sum += __expf(S[cidx] - mx);
    __syncthreads();
    red[row][tp] = sum;
    __syncthreads();
    sum = 0.0f;
#pragma unroll
    for (int i = 0; i < 16; ++i) sum += red[row][i];
    float inv = 1.0f / sum;
    for (int cidx = tp; cidx < SEQ; cidx += 16) {
        float p = __expf(S[cidx] - mx) * inv;
        S[cidx] = p;                 // normalized attn to d_out
        P[row][cidx] = f2bf(p);      // bf16 copy for P*V
    }
    __threadfence_block();
    __syncthreads();

    // phase 3: ctx[16x64] = P[16x1024] @ V ; waves 0-3 each own 16 cols
    if (wave < 4) {
        const bf16_t* Vbase = Vt + ((size_t)((b * HEADS + h) * DK + wave * 16)) * SEQ;
        v8f c = {};
        for (int k0 = 0; k0 < SEQ; k0 += 32) {
            v16bf a  = load_a_frag(&P[0][0] + k0, SEQ);   // ds loads
            v16bf bb = load_b_frag(Vbase + k0, SEQ);
            c = wmma_bf16(a, bb, c);
        }
        int n = wave * 16 + (lane & 15);
        int half = lane >> 4;
#pragma unroll
        for (int r = 0; r < 8; ++r) {
            int m = q0 + r + half * 8;
            ctxb[((size_t)(b * SEQ + m)) * DIM + h * DK + n] = f2bf(c[r]);
        }
    }
}

// ---------- launcher ----------

extern "C" void kernel_launch(void* const* d_in, const int* in_sizes, int n_in,
                              void* d_out, int out_size, void* d_ws, size_t ws_size,
                              hipStream_t stream) {
    const float* hidden = (const float*)d_in[0];
    const float* key    = (const float*)d_in[1];
    const float* cvects = (const float*)d_in[2];
    const float* Wq = (const float*)d_in[3];
    const float* bq = (const float*)d_in[4];
    const float* Wk = (const float*)d_in[5];
    const float* bk = (const float*)d_in[6];
    const float* Wv = (const float*)d_in[7];
    const float* bv = (const float*)d_in[8];
    const float* Wo = (const float*)d_in[9];
    const float* bo = (const float*)d_in[10];

    const size_t MD = (size_t)BATCH * SEQ * DIM;   // 8M activation elems
    const size_t WD = (size_t)DIM * DIM;           // 1M weight elems

    char* p = (char*)d_ws;
    bf16_t* hb   = (bf16_t*)p; p += MD * 2;
    bf16_t* kbin = (bf16_t*)p; p += MD * 2;
    bf16_t* cb   = (bf16_t*)p; p += MD * 2;
    bf16_t* wqT  = (bf16_t*)p; p += WD * 2;
    bf16_t* wkT  = (bf16_t*)p; p += WD * 2;
    bf16_t* wvT  = (bf16_t*)p; p += WD * 2;
    bf16_t* woT  = (bf16_t*)p; p += WD * 2;
    bf16_t* qb   = (bf16_t*)p; p += MD * 2;
    bf16_t* kbf  = (bf16_t*)p; p += MD * 2;
    bf16_t* vt   = (bf16_t*)p; p += MD * 2;
    bf16_t* ctxb = (bf16_t*)p; p += MD * 2;

    float* out  = (float*)d_out;            // [B, SEQ, DIM]
    float* attn = out + MD;                 // [B, H, SEQ, SEQ]

    // 1) convert activations + transpose-convert weights to bf16
    int cvtBlocks = (int)((MD + 255) / 256);
    cvt_f32_bf16_kernel<<<cvtBlocks, 256, 0, stream>>>(hidden, hb,   (int)MD);
    cvt_f32_bf16_kernel<<<cvtBlocks, 256, 0, stream>>>(key,    kbin, (int)MD);
    cvt_f32_bf16_kernel<<<cvtBlocks, 256, 0, stream>>>(cvects, cb,   (int)MD);
    int wBlocks = (int)(WD / 256);
    transpose_cvt_kernel<<<wBlocks, 256, 0, stream>>>(Wq, wqT);
    transpose_cvt_kernel<<<wBlocks, 256, 0, stream>>>(Wk, wkT);
    transpose_cvt_kernel<<<wBlocks, 256, 0, stream>>>(Wv, wvT);
    transpose_cvt_kernel<<<wBlocks, 256, 0, stream>>>(Wo, woT);

    // 2) Q/K/V projections (M=8192, N=K=1024); 128x128 block tiles
    int M = BATCH * SEQ;
    int gemmBlocks = (M / BM) * (DIM / BN);        // 512
    gemm_bf16_tiled_kernel<<<gemmBlocks, 256, 0, stream>>>(hb,   wqT, bq, nullptr, qb,  M, DIM, DIM, 0);
    gemm_bf16_tiled_kernel<<<gemmBlocks, 256, 0, stream>>>(kbin, wkT, bk, nullptr, kbf, M, DIM, DIM, 0);
    gemm_bf16_tiled_kernel<<<gemmBlocks, 256, 0, stream>>>(cb,   wvT, bv, nullptr, vt,  M, DIM, DIM, 1);

    // 3) fused attention: one block per (b, h, 16-row q-tile)
    attn_fused_kernel<<<BATCH * HEADS * (SEQ / 16), 256, 0, stream>>>(qb, kbf, vt, attn, ctxb);

    // 4) output projection -> fp32 d_out
    gemm_bf16_tiled_kernel<<<gemmBlocks, 256, 0, stream>>>(ctxb, woT, bo, out, nullptr, M, DIM, DIM, 2);
}
